// RepetHead_55697135894725
// MI455X (gfx1250) — matrix operand order repaired and checked by reading, not compiled, not run on hardware
//
#include <hip/hip_runtime.h>

#define Bn 4
#define Cn 256
#define Hn 64
#define Wn 64
#define HW 4096            // H*W
#define Kt 9               // kernel taps
#define OC 18              // offset channels = 2*K
#define KCH 72             // K-chunks of 32 over C*K = 2304
#define WT_PER_LAYER (16 * KCH * 32 * 16)   // 589824 bf16 elems per layer

typedef __attribute__((ext_vector_type(16))) __bf16 v16bf;
typedef __attribute__((ext_vector_type(8)))  float  v8f;

// ---------------------------------------------------------------------------
// Kernel 1: convert main conv weights fp32 -> bf16, pre-swizzled into the
// exact WMMA A-fragment layout: Wt[((r*16+mt)*72+q)*32+lane][e], so the GEMM
// kernel loads A as one contiguous 32B read per lane (global_load_b128 x2).
// A-matrix 16x32 bf16 layout: lane: M=lane%16, half=lane/16;
//   element e: K = 16*half + e (contiguous k order used here, same as B)
// Chunk q: tap kk = q/8, channel base (q%8)*32; global K = kk*256 + c.
// ---------------------------------------------------------------------------
__global__ __launch_bounds__(256) void convert_w_kernel(
    const float* __restrict__ w, __bf16* __restrict__ Wt, int total)
{
    int tid = blockIdx.x * 256 + threadIdx.x;
    if (tid >= total) return;
    int e    = tid & 15;
    int lane = (tid >> 4) & 31;
    int q    = (tid >> 9) % KCH;
    int mtr  = tid / (512 * KCH);      // r*16 + mt
    int mt   = mtr & 15;
    int r    = mtr >> 4;

    int o    = mt * 16 + (lane & 15);
    int half = lane >> 4;
    int v    = e >> 1, p = e & 1;
    int cc   = 16 * half + e;   // K within chunk
    int c    = (q & 7) * 32 + cc;
    int kk   = q >> 3;

    size_t src = ((((size_t)r * Cn + o) * Cn + c) * Kt) + kk;
    Wt[tid] = (__bf16)w[src];
}

// ---------------------------------------------------------------------------
// Kernel 2: 3x3 offset conv (C=256 -> 18), stride 1, pad 1, fp32 direct with
// LDS staging.  One block per (b,h) row: 288 threads = 18 oc x 16 w-slots,
// each thread produces 4 outputs (w, w+16, w+32, w+48).
// ---------------------------------------------------------------------------
__global__ __launch_bounds__(288) void offset_conv_kernel(
    const float* __restrict__ x, const float* __restrict__ ow,
    const float* __restrict__ ob, float* __restrict__ off)
{
    const int b = blockIdx.x / Hn;
    const int h = blockIdx.x % Hn;
    const int tid = threadIdx.x;
    const int oc = tid >> 4;          // 0..17
    const int wi = tid & 15;          // 0..15

    __shared__ float xs[3][66];       // 3 rows, halo cols [-1..64]
    __shared__ float ws_[OC * Kt];    // weights for current c

    float acc0 = 0.f, acc1 = 0.f, acc2 = 0.f, acc3 = 0.f;

    for (int c = 0; c < Cn; ++c) {
        __syncthreads();
        for (int i = tid; i < 198; i += 288) {
            int row = i / 66, col = i % 66;
            int yy = h - 1 + row, xx = col - 1;
            float val = 0.f;
            if (yy >= 0 && yy < Hn && xx >= 0 && xx < Wn)
                val = x[((b * Cn + c) * Hn + yy) * Wn + xx];
            xs[row][col] = val;
        }
        if (tid < OC * Kt) {
            int oc2 = tid / Kt, tap = tid % Kt;
            ws_[tid] = ow[(oc2 * Cn + c) * Kt + tap];
        }
        __syncthreads();

        float s0 = 0.f, s1 = 0.f, s2 = 0.f, s3 = 0.f;
#pragma unroll
        for (int tap = 0; tap < 9; ++tap) {
            int ky = tap / 3, kx = tap % 3;
            float wv = ws_[oc * Kt + tap];
            s0 = fmaf(xs[ky][wi +  0 + kx], wv, s0);
            s1 = fmaf(xs[ky][wi + 16 + kx], wv, s1);
            s2 = fmaf(xs[ky][wi + 32 + kx], wv, s2);
            s3 = fmaf(xs[ky][wi + 48 + kx], wv, s3);
        }
        acc0 += s0; acc1 += s1; acc2 += s2; acc3 += s3;
    }

    float bias = ob[oc];
    size_t base = ((size_t)(b * OC + oc) * Hn + h) * Wn;
    off[base + wi +  0] = acc0 + bias;
    off[base + wi + 16] = acc1 + bias;
    off[base + wi + 32] = acc2 + bias;
    off[base + wi + 48] = acc3 + bias;
}

// ---------------------------------------------------------------------------
// Kernel 3: fused bilinear-sample + WMMA GEMM + ReLU.
// Block = 256 threads (8 waves), covers a 32-pixel tile (two 16-px N-tiles,
// one row segment).  All 9 taps' bilinear corners precomputed to LDS once.
// Main loop over 72 K-chunks with double-buffered B fragments: per chunk each
// thread prefetch-samples 4 values for chunk q+1, each wave issues 4
// v_wmma_f32_16x16x32_bf16 (2 M-tiles x 2 N-tiles) on chunk q, then one
// barrier.  A fragments are reused across both N-tiles.
// ---------------------------------------------------------------------------
__global__ __launch_bounds__(256) void deform_gemm_kernel(
    const float* __restrict__ x, const float* __restrict__ off,
    const __bf16* __restrict__ Wt, float* __restrict__ y)
{
    const int tid  = threadIdx.x;
    const int lane = tid & 31;
    const int wave = tid >> 5;
    const int pbase = blockIdx.x * 32;     // 32 pixels, same b, same row
    const int b   = pbase >> 12;           // / 4096
    const int rem = pbase & 4095;
    const int h   = rem >> 6;
    const int w0  = rem & 63;

    __shared__ __align__(32) __bf16 Bf[2][2][32][16]; // [buf][ntile][lane][e]
    __shared__ int   sAddr[Kt][32][4];                // corner addrs (c=0 plane)
    __shared__ float sWgt[Kt][32][4];                 // corner weights

    // ---- precompute bilinear corners for all 9 taps x 32 pixels ----------
    for (int t = tid; t < Kt * 32; t += 256) {
        const int kk = t >> 5;
        const int n  = t & 31;
        const int ww = w0 + n;
        float dy = off[((b * OC + 2 * kk)     * Hn + h) * Wn + ww];
        float dx = off[((b * OC + 2 * kk + 1) * Hn + h) * Wn + ww];
        float py = (float)h  + (float)(kk / 3 - 1) + dy;
        float px = (float)ww + (float)(kk % 3 - 1) + dx;
        float fy = floorf(py), fx = floorf(px);
        int y0 = (int)fy, x0 = (int)fx;
        float wy = py - fy, wx = px - fx;
        float cw[4] = { (1.f - wy) * (1.f - wx), (1.f - wy) * wx,
                        wy * (1.f - wx),         wy * wx };
        int cy[4] = { y0, y0, y0 + 1, y0 + 1 };
        int cx[4] = { x0, x0 + 1, x0, x0 + 1 };
#pragma unroll
        for (int j = 0; j < 4; ++j) {
            bool valid = (cy[j] >= 0) && (cy[j] < Hn) &&
                         (cx[j] >= 0) && (cx[j] < Wn);
            int yc = min(max(cy[j], 0), Hn - 1);
            int xc = min(max(cx[j], 0), Wn - 1);
            sAddr[kk][n][j] = b * Cn * HW + yc * Wn + xc;
            sWgt[kk][n][j]  = valid ? cw[j] : 0.f;
        }
    }
    __syncthreads();

    // this thread's fixed role: pixel n, channel-pair ccp
    const int n   = tid & 31;
    const int nt  = n >> 4;          // N-tile 0/1
    const int nr  = n & 15;          // N within tile
    const int ccp = tid >> 5;        // 0..7 -> channels {2ccp, 2ccp+1, +16, +17}

    // sample 4 bf16 B-elements of chunk q for this thread
    auto sample4 = [&](int q, float& v0, float& v1, float& v2, float& v3) {
        const int kk = q >> 3;
        const int cb = (q & 7) * 32;
        const int a0 = sAddr[kk][n][0], a1 = sAddr[kk][n][1];
        const int a2 = sAddr[kk][n][2], a3 = sAddr[kk][n][3];
        const float f0 = sWgt[kk][n][0], f1 = sWgt[kk][n][1];
        const float f2 = sWgt[kk][n][2], f3 = sWgt[kk][n][3];
        const int c0 = (cb + 2 * ccp) * HW;           // cc = 2*ccp
        const int c1 = c0 + HW;                       // cc = 2*ccp+1
        const int c2 = c0 + 16 * HW;                  // cc = 2*ccp+16
        const int c3 = c2 + HW;                       // cc = 2*ccp+17
        v0 = x[a0+c0]*f0 + x[a1+c0]*f1 + x[a2+c0]*f2 + x[a3+c0]*f3;
        v1 = x[a0+c1]*f0 + x[a1+c1]*f1 + x[a2+c1]*f2 + x[a3+c1]*f3;
        v2 = x[a0+c2]*f0 + x[a1+c2]*f1 + x[a2+c2]*f2 + x[a3+c2]*f3;
        v3 = x[a0+c3]*f0 + x[a1+c3]*f1 + x[a2+c3]*f2 + x[a3+c3]*f3;
    };
    // store into B-fragment layout: row = 16*(cc/16)+nr, e = cc%16
    auto writeB = [&](int bufSel, float v0, float v1, float v2, float v3) {
        __bf16* p0 = &Bf[bufSel][nt][nr][2 * ccp];        // cc, cc+1  (half 0)
        p0[0] = (__bf16)v0; p0[1] = (__bf16)v1;
        __bf16* p1 = &Bf[bufSel][nt][16 + nr][2 * ccp];   // cc+16/17  (half 1)
        p1[0] = (__bf16)v2; p1[1] = (__bf16)v3;
    };

    v8f acc00 = {0.f,0.f,0.f,0.f,0.f,0.f,0.f,0.f};  // Mtile0 x Ntile0
    v8f acc01 = acc00, acc10 = acc00, acc11 = acc00;

    { // prologue: fill buffer 0 with chunk 0
        float v0, v1, v2, v3;
        sample4(0, v0, v1, v2, v3);
        writeB(0, v0, v1, v2, v3);
    }
    __syncthreads();

    for (int q = 0; q < KCH; ++q) {
        const int buf = q & 1;
        float v0, v1, v2, v3;
        const bool pre = (q < KCH - 1);
        if (pre) sample4(q + 1, v0, v1, v2, v3);   // overlap with WMMA below

        v16bf bf0 = *(const v16bf*)(&Bf[buf][0][lane][0]);
        v16bf bf1 = *(const v16bf*)(&Bf[buf][1][lane][0]);
        const v16bf af0 = *(const v16bf*)(Wt +
            ((((size_t)(wave * 2 + 0) * KCH + q) * 32 + lane) << 4));
        const v16bf af1 = *(const v16bf*)(Wt +
            ((((size_t)(wave * 2 + 1) * KCH + q) * 32 + lane) << 4));
        acc00 = __builtin_amdgcn_wmma_f32_16x16x32_bf16(
                    false, af0, false, bf0, (short)0, acc00, false, false);
        acc01 = __builtin_amdgcn_wmma_f32_16x16x32_bf16(
                    false, af0, false, bf1, (short)0, acc01, false, false);
        acc10 = __builtin_amdgcn_wmma_f32_16x16x32_bf16(
                    false, af1, false, bf0, (short)0, acc10, false, false);
        acc11 = __builtin_amdgcn_wmma_f32_16x16x32_bf16(
                    false, af1, false, bf1, (short)0, acc11, false, false);

        if (pre) writeB(1 - buf, v0, v1, v2, v3);  // other buffer: WAR-safe
        __syncthreads();                            // single barrier per chunk
    }

    // --- epilogue: ReLU + store. D layout: M = a + 8*(lane/16), N = lane%16
    const int en = lane & 15;
    const int halfSel = lane >> 4;
    const size_t pixBase = (size_t)b * Cn * HW + (size_t)h * Wn + w0;
#pragma unroll
    for (int a = 0; a < 8; ++a) {
        int o0 = wave * 32 + a + halfSel * 8;
        int o1 = o0 + 16;
        y[pixBase + en      + (size_t)o0 * HW] = fmaxf(acc00[a], 0.f);
        y[pixBase + 16 + en + (size_t)o0 * HW] = fmaxf(acc01[a], 0.f);
        y[pixBase + en      + (size_t)o1 * HW] = fmaxf(acc10[a], 0.f);
        y[pixBase + 16 + en + (size_t)o1 * HW] = fmaxf(acc11[a], 0.f);
    }
}

// ---------------------------------------------------------------------------
extern "C" void kernel_launch(void* const* d_in, const int* in_sizes, int n_in,
                              void* d_out, int out_size, void* d_ws, size_t ws_size,
                              hipStream_t stream) {
    const float* x_in  = (const float*)d_in[0];
    const float* off_w = (const float*)d_in[1];
    const float* off_b = (const float*)d_in[2];
    const float* w     = (const float*)d_in[3];
    float* out = (float*)d_out;

    char* ws = (char*)d_ws;
    float*  offs = (float*)ws;                                    // 4*18*64*64*4 = 1179648 B
    float*  x0   = (float*)(ws + 1179648);                        // 16.8 MB
    float*  x1   = x0 + (size_t)Bn * Cn * HW;                     // 16.8 MB
    __bf16* Wt   = (__bf16*)(x1 + (size_t)Bn * Cn * HW);          // 4.7 MB

    const int wt_total = 4 * WT_PER_LAYER;                        // 2359296
    convert_w_kernel<<<(wt_total + 255) / 256, 256, 0, stream>>>(w, Wt, wt_total);

    const float* cur = x_in;
    float* bufs[2] = { x0, x1 };
    for (int r = 0; r < 4; ++r) {
        offset_conv_kernel<<<Bn * Hn, 288, 0, stream>>>(
            cur, off_w + (size_t)r * OC * Cn * Kt, off_b + (size_t)r * OC, offs);
        float* nxt = (r == 3) ? out : bufs[r & 1];
        deform_gemm_kernel<<<(Bn * HW) / 32, 256, 0, stream>>>(
            cur, offs, Wt + (size_t)r * WT_PER_LAYER, nxt);
        cur = nxt;
    }
}
